// LSTMModel_39814346834405
// MI455X (gfx1250) — compile-verified, matrix operand order checked by reading, not run on hardware
//
#include <hip/hip_runtime.h>
#include <hip/hip_bf16.h>

typedef _Float16 v16h __attribute__((ext_vector_type(16)));
typedef _Float16 v8h  __attribute__((ext_vector_type(8)));
typedef float    v8f  __attribute__((ext_vector_type(8)));
typedef float    v4f  __attribute__((ext_vector_type(4)));

#define TT    168      // timesteps
#define PP    16       // input features
#define HH    24       // hidden size
#define NT    6        // N tiles of 16 covering 4H = 96 gate outputs
#define WAVES 4        // waves (16-row batch tiles) per block
#define GROW  100      // gate LDS buffer row stride in floats (96 cols + pad, breaks bank aliasing)
#define HROW  40       // h LDS buffer row stride in halfs (32 K-padded cols + pad; 80B keeps 16B align)

__device__ __forceinline__ float fast_rcp(float x) {
#if __has_builtin(__builtin_amdgcn_rcpf)
  return __builtin_amdgcn_rcpf(x);
#else
  return 1.0f / x;   // host-pass parse fallback; device pass uses v_rcp_f32
#endif
}

// CDNA5 has a hardware v_tanh_f32 (TANH listed as a 32-bit trans op).
// Use the builtin when declared; otherwise fall back to the exp/rcp form.
__device__ __forceinline__ float fast_tanh(float x) {
#if __has_builtin(__builtin_amdgcn_tanhf)
  return __builtin_amdgcn_tanhf(x);
#else
  float e = __expf(-2.0f * x);
  return (1.0f - e) * fast_rcp(1.0f + e);
#endif
}
__device__ __forceinline__ float fast_sigmoid(float x) {
#if __has_builtin(__builtin_amdgcn_tanhf)
  // sigmoid(x) = 0.5 + 0.5*tanh(x/2): mul + tanh + fma, no rcp chain
  return fmaf(0.5f, __builtin_amdgcn_tanhf(0.5f * x), 0.5f);
#else
  return fast_rcp(1.0f + __expf(-x));
#endif
}

__global__ __launch_bounds__(WAVES * 32, 1)
void lstm_fused_wmma(const float* __restrict__ x,
                     const float* __restrict__ W_ih,
                     const float* __restrict__ W_hh,
                     const float* __restrict__ b_ih,
                     const float* __restrict__ b_hh,
                     const float* __restrict__ W_lin,
                     const float* __restrict__ b_lin,
                     float* __restrict__ out,
                     int Btotal)
{
  // Per-wave private LDS buffers (intra-wave traffic only -> no barriers;
  // DS ops are in-order within a wave on CDNA5).
  __shared__ float gbuf[WAVES][16 * GROW];                       // 16 x 96 gate pre-activations (f32)
  __shared__ __align__(16) _Float16 hbuf[WAVES][16 * HROW];      // 16 x 32 h in WMMA-A f16 layout

  const int lane = threadIdx.x & 31;
  const int w    = threadIdx.x >> 5;
  const int tile = blockIdx.x * WAVES + w;
  const int rowb = tile * 16;
  if (rowb >= Btotal) return;          // wave-uniform: EXEC stays all-ones for WMMA

  const int nn = lane & 15;            // N (or M) index within a 16-wide tile
  const int hi = lane >> 4;            // lane half selects K chunk per WMMA A/B layout
  const int k0 = hi << 3;              // K chunk base: 0 (lanes 0-15) or 8 (lanes 16-31)

  // ---- pack weights into WMMA B-operand registers once (K zero-padded to 32) ----
  // B layout (16-bit, 16x16x32): lane nn = column N; elems 0..7 = K k0..k0+7,
  // elems 8..15 = K k0+16..k0+23 (zero where K >= actual K dim).
  v16h wih[NT], whh[NT];
  float bias[NT];
  #pragma unroll
  for (int n = 0; n < NT; ++n) {
    const int g = n * 16 + nn;         // gate row of W (g = x . W^T)
    v16h a = {};
    v16h b = {};
    #pragma unroll
    for (int j = 0; j < 8; ++j) {
      a[j] = (_Float16)W_ih[g * PP + k0 + j];            // K 0..15 valid, 16..31 pad (stay 0)
      b[j] = (_Float16)W_hh[g * HH + k0 + j];            // K 0..15 valid
      float w2 = W_hh[g * HH + 16 + j];                  // K 16..23 (lower-half lanes only)
      b[8 + j] = (_Float16)(hi ? 0.0f : w2);             // K 24..31 pad -> 0
    }
    wih[n] = a;
    whh[n] = b;
    bias[n] = b_ih[g] + b_hh[g];
  }

  // ---- zero h buffer: h0 = 0 and K-padding columns stay 0 forever ----
  for (int i = lane; i < 16 * HROW; i += 32) hbuf[w][i] = (_Float16)0.0f;

  // ---- per-lane ownership of the 16x24 cell state: 12 (row,col) elements ----
  int rowj[12], colj[12];
  float cstate[12];
  #pragma unroll
  for (int j = 0; j < 12; ++j) {
    int e = j * 32 + lane;             // 384 = 16*24 elements over 32 lanes
    rowj[j] = e / HH;
    colj[j] = e % HH;
    cstate[j] = 0.0f;
  }

  // x element (b,t,p) at ((b*TT)+t)*PP + p ; each lane reads 8 consecutive f32
  const float* xrow = x + ((size_t)(rowb + nn) * TT) * PP + k0;
  const _Float16* hA0 = &hbuf[w][nn * HROW + k0];        // A chunk0; chunk1 at +16 halfs

  #pragma unroll 1
  for (int t = 0; t < TT; ++t) {
    if (t + 4 < TT) __builtin_prefetch(xrow + (size_t)(t + 4) * PP, 0, 0);

    // ---- A operand for x_t (K=16, padded to 32) ----
    const v4f* xp = (const v4f*)(xrow + (size_t)t * PP);
    v4f x0 = xp[0];
    v4f x1 = xp[1];
    v16h ax = {};
    #pragma unroll
    for (int j = 0; j < 4; ++j) {
      ax[j]     = (_Float16)x0[j];
      ax[4 + j] = (_Float16)x1[j];
    }

    // ---- A operand for h_{t-1} straight from LDS (already in A layout) ----
    v8h h0 = *(const v8h*)(hA0);
    v8h h1 = *(const v8h*)(hA0 + 16);
    v16h ah = __builtin_shufflevector(h0, h1,
                0,1,2,3,4,5,6,7,8,9,10,11,12,13,14,15);

    // ---- gates = bias + x_t W_ih^T + h W_hh^T : 6 tiles x 2 WMMAs ----
    #pragma unroll
    for (int n = 0; n < NT; ++n) {
      v8f acc;
      #pragma unroll
      for (int v = 0; v < 8; ++v) acc[v] = bias[n];      // C row bias (depends only on N)
      acc = __builtin_amdgcn_wmma_f32_16x16x32_f16(false, ax, false, wih[n],
                                                   (short)0, acc, false, false);
      acc = __builtin_amdgcn_wmma_f32_16x16x32_f16(false, ah, false, whh[n],
                                                   (short)0, acc, false, false);
      // scatter C tile (VGPR v -> row v + 8*hi, col n*16+nn) to gate buffer
      #pragma unroll
      for (int v = 0; v < 8; ++v) {
        int r = v + (hi << 3);
        gbuf[w][r * GROW + n * 16 + nn] = acc[v];
      }
    }

    // ---- LSTM cell update: gates at cols {c, 24+c, 48+c, 72+c} of the row ----
    #pragma unroll
    for (int j = 0; j < 12; ++j) {
      const float* gp = &gbuf[w][rowj[j] * GROW + colj[j]];
      float iv = fast_sigmoid(gp[0]);
      float fv = fast_sigmoid(gp[24]);
      float gv = fast_tanh(gp[48]);
      float ov = fast_sigmoid(gp[72]);
      float c  = fv * cstate[j] + iv * gv;
      cstate[j] = c;
      hbuf[w][rowj[j] * HROW + colj[j]] = (_Float16)(ov * fast_tanh(c));
    }
  }

  // ---- head: out = tanh(h_last) @ W_lin^T + b_lin ----
  #pragma unroll
  for (int j = 0; j < 12; ++j) {
    float hv = (float)hbuf[w][rowj[j] * HROW + colj[j]];
    gbuf[w][rowj[j] * GROW + colj[j]] = fast_tanh(hv);   // stage tanh(h) per row
  }
  #pragma unroll 1
  for (int j = 0; j < 12; ++j) {
    const int row = rowj[j], oc = colj[j];
    float s = b_lin[oc];
    #pragma unroll
    for (int k = 0; k < HH; ++k)
      s += gbuf[w][row * GROW + k] * W_lin[oc * HH + k];
    out[(size_t)(rowb + row) * HH + oc] = s;
  }
}

extern "C" void kernel_launch(void* const* d_in, const int* in_sizes, int n_in,
                              void* d_out, int out_size, void* d_ws, size_t ws_size,
                              hipStream_t stream) {
  const float* x     = (const float*)d_in[0];
  const float* W_ih  = (const float*)d_in[1];
  const float* W_hh  = (const float*)d_in[2];
  const float* b_ih  = (const float*)d_in[3];
  const float* b_hh  = (const float*)d_in[4];
  const float* W_lin = (const float*)d_in[5];
  const float* b_lin = (const float*)d_in[6];
  float* out = (float*)d_out;
  (void)d_ws; (void)ws_size; (void)n_in; (void)out_size;

  const int Btotal = in_sizes[0] / (TT * PP);   // 8192
  const int tiles  = (Btotal + 15) / 16;        // 512 wave tiles
  const int blocks = (tiles + WAVES - 1) / WAVES;
  hipLaunchKernelGGL(lstm_fused_wmma, dim3(blocks), dim3(WAVES * 32), 0, stream,
                     x, W_ih, W_hh, b_ih, b_hh, W_lin, b_lin, out, Btotal);
}
